// biLM_26980984554222
// MI455X (gfx1250) — compile-verified, matrix operand order checked by reading, not run on hardware
//
#include <hip/hip_runtime.h>
#include <hip/hip_bf16.h>

// Problem sizes (fixed by the reference)
#define T_LEN 512
#define BATCH 32
#define HID   512
#define G4H   2048   // 4*HID

typedef __attribute__((ext_vector_type(16))) __bf16 v16bf;
typedef __attribute__((ext_vector_type(8)))  __bf16 v8bf;
typedef __attribute__((ext_vector_type(8)))  float  v8f;

__device__ __forceinline__ float sigmoidf_(float x) {
    return 1.0f / (1.0f + __expf(-x));
}

__device__ __forceinline__ v8f v8f_zero() {
    v8f z;
#pragma unroll
    for (int i = 0; i < 8; ++i) z[i] = 0.0f;
    return z;
}

// Load one 16-bit WMMA fragment (A 16x32 or B 32x16 layout) given a pointer to
// the lane's first K element. Per ISA: elements e0..7 = K base..base+7,
// e8..15 = K base+16..base+23  -> two 16-byte contiguous loads.
__device__ __forceinline__ v16bf load_frag(const __bf16* base) {
    v8bf lo = *(const v8bf*)(base);
    v8bf hi = *(const v8bf*)(base + 16);
    v16bf r;
#pragma unroll
    for (int e = 0; e < 8; ++e) { r[e] = lo[e]; r[e + 8] = hi[e]; }
    return r;
}

// ---------------------------------------------------------------------------
// Phase 0: zero the outputs+hidden regions of d_out (masked steps stay zero)
// ---------------------------------------------------------------------------
__global__ void zero4_kernel(float4* p, size_t n4) {
    size_t i = (size_t)blockIdx.x * blockDim.x + threadIdx.x;
    size_t stride = (size_t)gridDim.x * blockDim.x;
    float4 z; z.x = z.y = z.z = z.w = 0.0f;
    for (; i < n4; i += stride) p[i] = z;
}

// ---------------------------------------------------------------------------
// Phase 1: embedding gather -> f32 output (third return value) + bf16 copy for
// the WMMA input projection.
// ---------------------------------------------------------------------------
__global__ __launch_bounds__(128) void embed_kernel(
    const int* __restrict__ seq, const float* __restrict__ emb,
    float* __restrict__ emb_out, __bf16* __restrict__ xbf)
{
    const int row = blockIdx.x;                 // row = t*BATCH + b
    const int tok = seq[row];
    const float4 v = ((const float4*)(emb + (size_t)tok * HID))[threadIdx.x];
    ((float4*)(emb_out + (size_t)row * HID))[threadIdx.x] = v;
    __bf16* xd = xbf + (size_t)row * HID + threadIdx.x * 4;
    xd[0] = (__bf16)v.x; xd[1] = (__bf16)v.y;
    xd[2] = (__bf16)v.z; xd[3] = (__bf16)v.w;
}

// ---------------------------------------------------------------------------
// Phase 2: convert the four weight matrices to bf16 (each [4H, H] row-major,
// rows contiguous in K -> directly usable as WMMA B fragments).
// ---------------------------------------------------------------------------
__global__ void cvt_kernel(const float* __restrict__ a, const float* __restrict__ b,
                           const float* __restrict__ c, const float* __restrict__ d,
                           __bf16* __restrict__ oa, __bf16* __restrict__ ob,
                           __bf16* __restrict__ oc, __bf16* __restrict__ od, int n)
{
    int i = blockIdx.x * blockDim.x + threadIdx.x;
    int stride = gridDim.x * blockDim.x;
    for (; i < n; i += stride) {
        oa[i] = (__bf16)a[i]; ob[i] = (__bf16)b[i];
        oc[i] = (__bf16)c[i]; od[i] = (__bf16)d[i];
    }
}

// ---------------------------------------------------------------------------
// Phase 3: input projections. XW[dir] = X @ W_ih[dir].T + (b_ih + b_hh)
// M = T*B = 16384, K = 512, N = 2048.  Each wave: 32x32 output (2x2 WMMA tiles).
// Block 256 threads = 8 waves covering 256 columns; grid (8, 512, 2).
// ---------------------------------------------------------------------------
__global__ __launch_bounds__(256) void gemm_xw_kernel(
    const __bf16* __restrict__ X,
    const __bf16* __restrict__ Wih_f, const __bf16* __restrict__ Wih_b,
    const float* __restrict__ bih_f, const float* __restrict__ bhh_f,
    const float* __restrict__ bih_b, const float* __restrict__ bhh_b,
    float* __restrict__ XWf, float* __restrict__ XWb)
{
    const int dir = blockIdx.z;
    const __bf16* W = dir ? Wih_b : Wih_f;
    const float* bi = dir ? bih_b : bih_f;
    const float* bh = dir ? bhh_b : bhh_f;
    float* XW = dir ? XWb : XWf;

    const int lane = threadIdx.x & 31;
    const int wave = threadIdx.x >> 5;
    const int half = lane >> 4;
    const int nloc = lane & 15;
    const int row0 = blockIdx.y * 32;
    const int col0 = blockIdx.x * 256 + wave * 32;

    v8f acc[2][2];
#pragma unroll
    for (int mt = 0; mt < 2; ++mt)
#pragma unroll
        for (int nt = 0; nt < 2; ++nt) acc[mt][nt] = v8f_zero();

    for (int kk = 0; kk < HID; kk += 32) {
        v16bf a[2], b[2];
#pragma unroll
        for (int mt = 0; mt < 2; ++mt) {
            const int m = row0 + mt * 16 + nloc;
            a[mt] = load_frag(X + (size_t)m * HID + kk + half * 8);
        }
#pragma unroll
        for (int nt = 0; nt < 2; ++nt) {
            const int n = col0 + nt * 16 + nloc;
            b[nt] = load_frag(W + (size_t)n * HID + kk + half * 8);
        }
#pragma unroll
        for (int mt = 0; mt < 2; ++mt)
#pragma unroll
            for (int nt = 0; nt < 2; ++nt)
                acc[mt][nt] = __builtin_amdgcn_wmma_f32_16x16x32_bf16(
                    false, a[mt], false, b[nt], (short)0, acc[mt][nt], false, false);
    }

#pragma unroll
    for (int nt = 0; nt < 2; ++nt) {
        const int n = col0 + nt * 16 + nloc;
        const float bias = bi[n] + bh[n];
#pragma unroll
        for (int mt = 0; mt < 2; ++mt) {
#pragma unroll
            for (int r = 0; r < 8; ++r) {
                const int m = row0 + mt * 16 + half * 8 + r;
                XW[(size_t)m * G4H + n] = acc[mt][nt][r] + bias;
            }
        }
    }
}

// ---------------------------------------------------------------------------
// Phase 4: recurrent scan. One persistent 1024-thread workgroup per direction.
// h: [32,512] bf16 in LDS (32 KB). Wave w owns h columns [16w, 16w+16) and the
// matching i/f/g/o gate tiles, so the nonlinear update is register-resident.
// c stays in registers for all 512 steps. Masked rows freeze by skipping stores.
// ---------------------------------------------------------------------------
__global__ __launch_bounds__(1024) void lstm_rec_kernel(
    const float* __restrict__ XWf, const float* __restrict__ XWb,
    const __bf16* __restrict__ Whh_f, const __bf16* __restrict__ Whh_b,
    const int* __restrict__ lengths,
    float* __restrict__ outs,   // [T,B,2,H]
    float* __restrict__ hid)    // [2,T,B,H]
{
    const int dir = blockIdx.x;
    const float*  XW = dir ? XWb : XWf;
    const __bf16* W  = dir ? Whh_b : Whh_f;

    __shared__ __bf16 hbuf[BATCH][HID];   // 32 KB

    const int lane = threadIdx.x & 31;
    const int wave = threadIdx.x >> 5;    // 0..31
    const int half = lane >> 4;
    const int nloc = lane & 15;
    const int hcol = wave * 16 + nloc;    // h column owned by this lane

    for (int i = threadIdx.x; i < BATCH * HID; i += blockDim.x)
        ((__bf16*)hbuf)[i] = (__bf16)0.0f;

    int lenf[2][8];
#pragma unroll
    for (int mt = 0; mt < 2; ++mt)
#pragma unroll
        for (int r = 0; r < 8; ++r)
            lenf[mt][r] = lengths[mt * 16 + half * 8 + r];

    float cfrag[2][8];
#pragma unroll
    for (int mt = 0; mt < 2; ++mt)
#pragma unroll
        for (int r = 0; r < 8; ++r) cfrag[mt][r] = 0.0f;

    __syncthreads();

    for (int t = 0; t < T_LEN; ++t) {
        v8f acc[2][4];
#pragma unroll
        for (int mt = 0; mt < 2; ++mt)
#pragma unroll
            for (int g = 0; g < 4; ++g) acc[mt][g] = v8f_zero();

        // gates += h @ W_hh.T   (K = 512, 16 k-steps of 32)
        for (int kk = 0; kk < HID; kk += 32) {
            v16bf a[2];
#pragma unroll
            for (int mt = 0; mt < 2; ++mt)
                a[mt] = load_frag(&hbuf[mt * 16 + nloc][kk + half * 8]);
#pragma unroll
            for (int g = 0; g < 4; ++g) {
                const int n = g * HID + wave * 16 + nloc;
                const v16bf b = load_frag(W + (size_t)n * HID + kk + half * 8);
#pragma unroll
                for (int mt = 0; mt < 2; ++mt)
                    acc[mt][g] = __builtin_amdgcn_wmma_f32_16x16x32_bf16(
                        false, a[mt], false, b, (short)0, acc[mt][g], false, false);
            }
        }

        __syncthreads();   // all waves finished reading hbuf for this step

#pragma unroll
        for (int mt = 0; mt < 2; ++mt) {
#pragma unroll
            for (int r = 0; r < 8; ++r) {
                const int b   = mt * 16 + half * 8 + r;   // batch row (C/D layout)
                const int len = lenf[mt][r];
                const bool valid = (t < len);
                int tt = dir ? (len - 1 - t) : t;         // source time for XW
                if (tt < 0) tt = 0;
                const float* xw = XW + ((size_t)tt * BATCH + b) * G4H;
                const float pi = acc[mt][0][r] + xw[0 * HID + hcol];
                const float pf = acc[mt][1][r] + xw[1 * HID + hcol];
                const float pg = acc[mt][2][r] + xw[2 * HID + hcol];
                const float po = acc[mt][3][r] + xw[3 * HID + hcol];
                const float ig = sigmoidf_(pi);
                const float fg = sigmoidf_(pf);
                const float gg = tanhf(pg);
                const float og = sigmoidf_(po);
                const float cn = fg * cfrag[mt][r] + ig * gg;
                const float hn = og * tanhf(cn);
                if (valid) {
                    cfrag[mt][r] = cn;
                    hbuf[b][hcol] = (__bf16)hn;
                    const int tout = dir ? tt : t;        // original timestep
                    outs[(((size_t)tout * BATCH + b) * 2 + dir) * HID + hcol] = hn;
                    hid[(((size_t)dir * T_LEN + tout) * BATCH + b) * HID + hcol] = hn;
                }
            }
        }

        __syncthreads();   // hbuf updates visible before next step's reads
    }
}

// ---------------------------------------------------------------------------
extern "C" void kernel_launch(void* const* d_in, const int* in_sizes, int n_in,
                              void* d_out, int out_size, void* d_ws, size_t ws_size,
                              hipStream_t stream) {
    const int*   seq   = (const int*)d_in[0];
    const int*   lens  = (const int*)d_in[1];
    const float* emb   = (const float*)d_in[2];
    const float* Wf_ih = (const float*)d_in[3];
    const float* Wf_hh = (const float*)d_in[4];
    const float* bf_ih = (const float*)d_in[5];
    const float* bf_hh = (const float*)d_in[6];
    const float* Wb_ih = (const float*)d_in[7];
    const float* Wb_hh = (const float*)d_in[8];
    const float* bb_ih = (const float*)d_in[9];
    const float* bb_hh = (const float*)d_in[10];

    float* out  = (float*)d_out;
    float* outs = out;                                        // [T,B,2,H]
    float* hid  = out + (size_t)T_LEN * BATCH * 2 * HID;      // [2,T,B,H]
    float* embo = hid + (size_t)2 * T_LEN * BATCH * HID;      // [T,B,H]

    // Workspace layout
    char* w = (char*)d_ws;
    __bf16* Xbf  = (__bf16*)w; w += (size_t)T_LEN * BATCH * HID * sizeof(__bf16); // 16 MB
    __bf16* Wihf = (__bf16*)w; w += (size_t)G4H * HID * sizeof(__bf16);           // 2 MB
    __bf16* Wihb = (__bf16*)w; w += (size_t)G4H * HID * sizeof(__bf16);
    __bf16* Whhf = (__bf16*)w; w += (size_t)G4H * HID * sizeof(__bf16);
    __bf16* Whhb = (__bf16*)w; w += (size_t)G4H * HID * sizeof(__bf16);
    float*  XWf  = (float*)w;  w += (size_t)T_LEN * BATCH * G4H * sizeof(float);  // 128 MB
    float*  XWb  = (float*)w;

    // Phase 0: zero outputs + hidden (2 * 16777216 floats)
    const size_t nzero4 = ((size_t)2 * T_LEN * BATCH * 2 * HID) / 4;
    zero4_kernel<<<2048, 256, 0, stream>>>((float4*)out, nzero4);

    // Phase 1: embedding gather + bf16 copy
    embed_kernel<<<T_LEN * BATCH, 128, 0, stream>>>(seq, emb, embo, Xbf);

    // Phase 2: weights -> bf16
    cvt_kernel<<<2048, 256, 0, stream>>>(Wf_ih, Wb_ih, Wf_hh, Wb_hh,
                                         Wihf, Wihb, Whhf, Whhb, G4H * HID);

    // Phase 3: input projections (both directions), biases folded in
    dim3 ggrid(G4H / 256, (T_LEN * BATCH) / 32, 2);
    gemm_xw_kernel<<<ggrid, 256, 0, stream>>>(Xbf, Wihf, Wihb,
                                              bf_ih, bf_hh, bb_ih, bb_hh,
                                              XWf, XWb);

    // Phase 4: recurrent scan, one persistent workgroup per direction
    lstm_rec_kernel<<<2, 1024, 0, stream>>>(XWf, XWb, Whhf, Whhb, lens, outs, hid);
}
